// MultiHeadAttention_18081812316469
// MI455X (gfx1250) — matrix operand order repaired and checked by reading, not compiled, not run on hardware
//
#include <hip/hip_runtime.h>

#define NHEADS 16
#define DK     64
#define DM     1024
#define SEQ    2048
#define BATCH  2

typedef __attribute__((ext_vector_type(16))) _Float16 v16h;
typedef __attribute__((ext_vector_type(8)))  float    v8f;
typedef __attribute__((ext_vector_type(4)))  unsigned v4u;
typedef __attribute__((ext_vector_type(8)))  unsigned v8u;

// ---------------------------------------------------------------------------
// Tensor Data Mover: async 2-D tile (tileW x tileH elements, f16) from global
// row-major (leading dim ldElems) into contiguous LDS. Issued by one wave,
// completion via TENSORcnt (in-order per wave -> double-buffer with wait<=N).
// D# layout per CDNA5 ISA.
// ---------------------------------------------------------------------------
__device__ __forceinline__ void tdm_load_2d_f16(void* ldsDst, const void* gsrc,
                                                unsigned tileW, unsigned tileH,
                                                unsigned ldElems) {
  unsigned lds_off = (unsigned)(size_t)ldsDst;          // low 32 bits = LDS offset
  unsigned long long ga = (unsigned long long)(size_t)gsrc;
  v4u g0;
  g0[0] = 1u;                                            // count=1 (valid), no gather
  g0[1] = lds_off;                                       // lds_addr
  g0[2] = (unsigned)ga;                                  // global_addr[31:0]
  g0[3] = ((unsigned)(ga >> 32) & 0x01FFFFFFu) | (2u << 30);  // addr[56:32] | type=2
  const unsigned dim0 = ldElems;                         // tensor_dim0 (elements)
  const unsigned dim1 = 0x400000u;                       // large: no OOB clamp
  v8u g1;
  g1[0] = (1u << 16);                                    // data_size=1 -> 2 bytes
  g1[1] = (dim0 & 0xFFFFu) << 16;                        // tensor_dim0[15:0]
  g1[2] = (dim0 >> 16) | ((dim1 & 0xFFFFu) << 16);       // dim0[31:16] | dim1[15:0]
  g1[3] = ((dim1 >> 16) & 0xFFFFu) | (tileW << 16);      // dim1[31:16] | tile_dim0
  g1[4] = tileH;                                         // tile_dim1 (tile_dim2=0)
  g1[5] = ldElems;                                       // tensor_dim0_stride[31:0]
  g1[6] = 0u;                                            // stride[47:32] | d1stride lo
  g1[7] = 0u;
  asm volatile("tensor_load_to_lds %0, %1" :: "s"(g0), "s"(g1) : "memory");
}

__device__ __forceinline__ void tdm_wait0() {
  asm volatile("s_wait_tensorcnt 0x0" ::: "memory");
}
__device__ __forceinline__ void tdm_wait1() {
  asm volatile("s_wait_tensorcnt 0x1" ::: "memory");
}
__device__ __forceinline__ void tdm_wait2() {
  asm volatile("s_wait_tensorcnt 0x2" ::: "memory");
}

// ---------------------------------------------------------------------------
// WMMA 16x16x32 f16 fragment helpers (wave32), reading from LDS tiles.
// A (16x32 MxK): lane L row m=L%16, half=L/16; elem i -> k=half*16+i.
// B (32x16 KxN): lane L col n=L%16, k = (L/16)*16 + i.
// C/D (16x16 f32): lane L col n=L%16, VGPR r -> row m=(L/16)*8+r.
// ---------------------------------------------------------------------------
template <int STRIDE>
__device__ __forceinline__ v16h frag_a_lds(const _Float16* Al, int rowBase) {
  int lane = threadIdx.x & 31;
  const _Float16* p = Al + (size_t)(rowBase + (lane & 15)) * STRIDE + ((lane >> 4) * 16);
  v16h a;
#pragma unroll
  for (int i = 0; i < 8; ++i)  a[i] = p[i];
#pragma unroll
  for (int i = 8; i < 16; ++i) a[i] = p[i];
  return a;
}

// B tile stored transposed in LDS: Bt[n][k], contiguous in k.
template <int STRIDE>
__device__ __forceinline__ v16h frag_b_bt(const _Float16* Bt, int colBase) {
  int lane = threadIdx.x & 31;
  const _Float16* p = Bt + (size_t)(colBase + (lane & 15)) * STRIDE + ((lane >> 4) * 16);
  v16h b;
#pragma unroll
  for (int i = 0; i < 16; ++i) b[i] = p[i];
  return b;
}

// B tile stored row-major in LDS: Br[k][64] (TDM-landed natural orientation).
__device__ __forceinline__ v16h frag_b_brow(const _Float16* Br, int colBase) {
  int lane = threadIdx.x & 31;
  const _Float16* p = Br + (size_t)((lane >> 4) * 16) * 64 + colBase + (lane & 15);
  v16h b;
#pragma unroll
  for (int i = 0; i < 16; ++i) b[i] = p[(size_t)i * 64];
  return b;
}

__device__ __forceinline__ void store_c_f32(float* __restrict__ C, int ldc,
                                            int row, int col, v8f c, float scale) {
  int lane = threadIdx.x & 31;
  int n  = col + (lane & 15);
  int m0 = row + (lane >> 4) * 8;
#pragma unroll
  for (int r = 0; r < 8; ++r) C[(size_t)(m0 + r) * ldc + n] = c[r] * scale;
}

__device__ __forceinline__ void store_c_f16(_Float16* __restrict__ C, int ldc,
                                            int row, int col, v8f c) {
  int lane = threadIdx.x & 31;
  int n  = col + (lane & 15);
  int m0 = row + (lane >> 4) * 8;
#pragma unroll
  for (int r = 0; r < 8; ++r) C[(size_t)(m0 + r) * ldc + n] = (_Float16)c[r];
}

// ---------------------------------------------------------------------------
// LDS staging for f32 sources (convert to f16 on the fly). 128 threads.
// A tile: 64 rows x 32 k, padded stride 40 (bank-conflict-free b128 reads).
// ---------------------------------------------------------------------------
__device__ __forceinline__ void stage_a_f32(const float* __restrict__ src, int ld,
                                            _Float16 (*dst)[40], int kb) {
  int t = threadIdx.x;
  int r = t >> 1, c0 = (t & 1) * 16;
  const float* p = src + (size_t)r * ld + kb + c0;
  __builtin_prefetch(p + 32, 0, 1);            // next k-tile
#pragma unroll
  for (int j = 0; j < 16; j += 4) {
    float4 v = *(const float4*)(p + j);
    dst[r][c0 + j + 0] = (_Float16)v.x;
    dst[r][c0 + j + 1] = (_Float16)v.y;
    dst[r][c0 + j + 2] = (_Float16)v.z;
    dst[r][c0 + j + 3] = (_Float16)v.w;
  }
}

// B tile from row-major f32 B[k][n] (32 x 64), stored transposed: Bt[n][k].
__device__ __forceinline__ void stage_bT_f32(const float* __restrict__ src, int ld,
                                             _Float16 (*dst)[40]) {
  int t = threadIdx.x;
  int k = t >> 2, n0 = (t & 3) * 16;
  const float* p = src + (size_t)k * ld + n0;
  __builtin_prefetch(p + (size_t)32 * ld, 0, 1);
#pragma unroll
  for (int j = 0; j < 16; j += 4) {
    float4 v = *(const float4*)(p + j);
    dst[n0 + j + 0][k] = (_Float16)v.x;
    dst[n0 + j + 1][k] = (_Float16)v.y;
    dst[n0 + j + 2][k] = (_Float16)v.z;
    dst[n0 + j + 3][k] = (_Float16)v.w;
  }
}

// ---------------------------------------------------------------------------
// Kernels. All GEMM blocks: 128 threads (4 waves), 64x64 output tile,
// wave w owns rows [w*16, w*16+16), 4 accumulators across N.
// ---------------------------------------------------------------------------

// C(f16)[4096,1024] = A(f32) @ B(f32)  — Q/K/V projections
__global__ void k_proj_gemm(const float* __restrict__ A, const float* __restrict__ B,
                            _Float16* __restrict__ C) {
  __shared__ _Float16 Al[64][40];
  __shared__ _Float16 Btl[64][40];
  const int col0 = blockIdx.x * 64;
  const int row0 = blockIdx.y * 64;
  const int w = threadIdx.x >> 5;
  v8f acc[4] = {};
  for (int kb = 0; kb < DM; kb += 32) {
    __syncthreads();
    stage_a_f32(A + (size_t)row0 * DM, DM, Al, kb);
    stage_bT_f32(B + (size_t)kb * DM + col0, DM, Btl);
    __syncthreads();
    v16h af = frag_a_lds<40>(&Al[0][0], w * 16);
#pragma unroll
    for (int c = 0; c < 4; ++c) {
      v16h bf = frag_b_bt<40>(&Btl[0][0], c * 16);
      acc[c] = __builtin_amdgcn_wmma_f32_16x16x32_f16(false, af, false, bf,
                                                      (short)0, acc[c], false, false);
    }
  }
#pragma unroll
  for (int c = 0; c < 4; ++c)
    store_c_f16(C, DM, row0 + w * 16, col0 + c * 16, acc[c]);
}

// scores[b,h,q,k] = (Q . K) / 8, raw pre-softmax.
// A and B^T tiles via TDM, double-buffered across the two k-steps.
__global__ void k_scores(const _Float16* __restrict__ Qf,
                         const _Float16* __restrict__ Kf,
                         float* __restrict__ attn) {
  __shared__ _Float16 Al[2][64][32];
  __shared__ _Float16 Btl[2][64][32];
  const int col0 = blockIdx.x * 64;
  const int row0 = blockIdx.y * 64;
  const int bh = blockIdx.z;
  const int b = bh >> 4, h = bh & 15;
  const _Float16* Qb = Qf + (size_t)b * SEQ * DM + h * DK;
  const _Float16* Kb = Kf + (size_t)b * SEQ * DM + h * DK;
  const int w = threadIdx.x >> 5;

  if (threadIdx.x < 32) {          // prologue: stage 0 in flight
    tdm_load_2d_f16(&Al[0][0][0],  Qb + (size_t)row0 * DM, 32, 64, DM);
    tdm_load_2d_f16(&Btl[0][0][0], Kb + (size_t)col0 * DM, 32, 64, DM);
  }

  v8f acc[4] = {};
#pragma unroll
  for (int i = 0; i < 2; ++i) {    // DK/32 = 2 k-steps
    __syncthreads();
    if (threadIdx.x < 32) {
      if (i == 0) {                // issue stage 1, then wait for stage 0 only
        tdm_load_2d_f16(&Al[1][0][0],  Qb + (size_t)row0 * DM + 32, 32, 64, DM);
        tdm_load_2d_f16(&Btl[1][0][0], Kb + (size_t)col0 * DM + 32, 32, 64, DM);
        tdm_wait2();
      } else {
        tdm_wait0();
      }
    }
    __syncthreads();
    v16h af = frag_a_lds<32>(&Al[i][0][0], w * 16);
#pragma unroll
    for (int c = 0; c < 4; ++c) {
      v16h bf = frag_b_bt<32>(&Btl[i][0][0], c * 16);
      acc[c] = __builtin_amdgcn_wmma_f32_16x16x32_f16(false, af, false, bf,
                                                      (short)0, acc[c], false, false);
    }
  }
  float* outp = attn + (size_t)bh * SEQ * SEQ;
#pragma unroll
  for (int c = 0; c < 4; ++c)
    store_c_f32(outp, SEQ, row0 + w * 16, col0 + c * 16, acc[c], 0.125f);
}

// In-place masked softmax over each length-SEQ row.
__global__ void mha_softmax(float* __restrict__ attn,
                            const unsigned char* __restrict__ mask) {
  int rowid = blockIdx.x;
  int q  = rowid % SEQ;
  int bh = rowid / SEQ;
  int b  = bh / NHEADS;
  float* rowp = attn + (size_t)rowid * SEQ;
  const unsigned char* mrow = mask + ((size_t)b * SEQ + q) * SEQ;

  __shared__ float red[256];
  int t = threadIdx.x;

  float lmax = -3.4e38f;
  for (int k = t; k < SEQ; k += 256) {
    float v = mrow[k] ? -1e9f : rowp[k];
    rowp[k] = v;
    lmax = fmaxf(lmax, v);
  }
  red[t] = lmax; __syncthreads();
  for (int s = 128; s > 0; s >>= 1) {
    if (t < s) red[t] = fmaxf(red[t], red[t + s]);
    __syncthreads();
  }
  float m = red[0]; __syncthreads();

  float lsum = 0.f;
  for (int k = t; k < SEQ; k += 256) {
    float e = __expf(rowp[k] - m);
    rowp[k] = e;
    lsum += e;
  }
  red[t] = lsum; __syncthreads();
  for (int s = 128; s > 0; s >>= 1) {
    if (t < s) red[t] += red[t + s];
    __syncthreads();
  }
  float inv = 1.0f / red[0];
  for (int k = t; k < SEQ; k += 256) rowp[k] *= inv;
}

// context = attn @ V.  A: f32 staged+cvt each step; B: V f16 via TDM,
// double-buffered so the DMA overlaps the A staging + WMMA of the prior step.
__global__ void k_context(const float* __restrict__ attn,
                          const _Float16* __restrict__ Vf,
                          _Float16* __restrict__ Ctx) {
  __shared__ _Float16 Al[64][40];
  __shared__ _Float16 Brl[2][32][64];
  const int row0 = blockIdx.y * 64;      // query rows; N tile = all 64 of d_k
  const int bh = blockIdx.z;
  const int b = bh >> 4, h = bh & 15;
  const float*    Ab = attn + (size_t)bh * SEQ * SEQ;
  const _Float16* Vb = Vf + (size_t)b * SEQ * DM + h * DK;
  const int w = threadIdx.x >> 5;

  if (threadIdx.x < 32)
    tdm_load_2d_f16(&Brl[0][0][0], Vb, 64, 32, DM);

  v8f acc[4] = {};
  for (int kb = 0; kb < SEQ; kb += 32) {
    const int buf = (kb >> 5) & 1;
    __syncthreads();                     // prior compute done: ping buffer free
    if (threadIdx.x < 32) {
      if (kb + 32 < SEQ) {
        tdm_load_2d_f16(&Brl[buf ^ 1][0][0], Vb + (size_t)(kb + 32) * DM, 64, 32, DM);
        tdm_wait1();                     // current tile landed, next in flight
      } else {
        tdm_wait0();
      }
    }
    stage_a_f32(Ab + (size_t)row0 * SEQ, SEQ, Al, kb);
    __syncthreads();
    v16h af = frag_a_lds<40>(&Al[0][0], w * 16);
#pragma unroll
    for (int c = 0; c < 4; ++c) {
      v16h bf = frag_b_brow(&Brl[buf][0][0], c * 16);
      acc[c] = __builtin_amdgcn_wmma_f32_16x16x32_f16(false, af, false, bf,
                                                      (short)0, acc[c], false, false);
    }
  }
  _Float16* Cb = Ctx + (size_t)b * SEQ * DM + h * DK;
#pragma unroll
  for (int c = 0; c < 4; ++c)
    store_c_f16(Cb, DM, row0 + w * 16, c * 16, acc[c]);
}

// proj(f32) = Ctx(f16) @ W_fc(f32).  A via TDM (double-buffered), B staged.
__global__ void k_outproj(const _Float16* __restrict__ Ctx,
                          const float* __restrict__ Wfc,
                          float* __restrict__ Pout) {
  __shared__ _Float16 Al[2][64][32];
  __shared__ _Float16 Btl[64][40];
  const int col0 = blockIdx.x * 64;
  const int row0 = blockIdx.y * 64;
  const int w = threadIdx.x >> 5;

  if (threadIdx.x < 32)
    tdm_load_2d_f16(&Al[0][0][0], Ctx + (size_t)row0 * DM, 32, 64, DM);

  v8f acc[4] = {};
  for (int kb = 0; kb < DM; kb += 32) {
    const int buf = (kb >> 5) & 1;
    __syncthreads();
    if (threadIdx.x < 32) {
      if (kb + 32 < DM) {
        tdm_load_2d_f16(&Al[buf ^ 1][0][0], Ctx + (size_t)row0 * DM + kb + 32,
                        32, 64, DM);
        tdm_wait1();
      } else {
        tdm_wait0();
      }
    }
    stage_bT_f32(Wfc + (size_t)kb * DM + col0, DM, Btl);
    __syncthreads();
    v16h af = frag_a_lds<32>(&Al[buf][0][0], w * 16);
#pragma unroll
    for (int c = 0; c < 4; ++c) {
      v16h bf = frag_b_bt<40>(&Btl[0][0], c * 16);
      acc[c] = __builtin_amdgcn_wmma_f32_16x16x32_f16(false, af, false, bf,
                                                      (short)0, acc[c], false, false);
    }
  }
#pragma unroll
  for (int c = 0; c < 4; ++c)
    store_c_f32(Pout, DM, row0 + w * 16, col0 + c * 16, acc[c], 1.0f);
}

// out = layer_norm(proj + input_Q)
__global__ void mha_ln_residual(const float* __restrict__ proj,
                                const float* __restrict__ inQ,
                                float* __restrict__ out) {
  int rowid = blockIdx.x;
  const float* p = proj + (size_t)rowid * DM;
  const float* q = inQ  + (size_t)rowid * DM;
  float* o = out + (size_t)rowid * DM;

  __shared__ float red[256];
  int t = threadIdx.x;

  float ls = 0.f;
  for (int i = t; i < DM; i += 256) ls += p[i] + q[i];
  red[t] = ls; __syncthreads();
  for (int s = 128; s > 0; s >>= 1) { if (t < s) red[t] += red[t + s]; __syncthreads(); }
  float mu = red[0] * (1.0f / DM); __syncthreads();

  float lv = 0.f;
  for (int i = t; i < DM; i += 256) {
    float d = p[i] + q[i] - mu;
    lv += d * d;
  }
  red[t] = lv; __syncthreads();
  for (int s = 128; s > 0; s >>= 1) { if (t < s) red[t] += red[t + s]; __syncthreads(); }
  float inv = rsqrtf(red[0] * (1.0f / DM) + 1e-5f);

  for (int i = t; i < DM; i += 256) o[i] = (p[i] + q[i] - mu) * inv;
}

// ---------------------------------------------------------------------------
// Launch
// ---------------------------------------------------------------------------
extern "C" void kernel_launch(void* const* d_in, const int* in_sizes, int n_in,
                              void* d_out, int out_size, void* d_ws, size_t ws_size,
                              hipStream_t stream) {
  const float* inQ  = (const float*)d_in[0];
  const float* inK  = (const float*)d_in[1];
  const float* inV  = (const float*)d_in[2];
  const unsigned char* mask = (const unsigned char*)d_in[3];
  const float* WQ   = (const float*)d_in[4];
  const float* WK   = (const float*)d_in[5];
  const float* WV   = (const float*)d_in[6];
  const float* Wfc  = (const float*)d_in[7];

  const size_t rowsTot  = (size_t)BATCH * SEQ;   // 4096
  const size_t actElems = rowsTot * DM;          // 4M

  float* out_norm = (float*)d_out;               // [B,S,DM]
  float* out_attn = (float*)d_out + actElems;    // [B,H,S,S]

  _Float16* Qf   = (_Float16*)d_ws;
  _Float16* Kf   = Qf  + actElems;
  _Float16* Vf   = Kf  + actElems;
  _Float16* Ctx  = Vf  + actElems;
  float*    proj = (float*)(Ctx + actElems);

  dim3 blk(128, 1, 1);                           // 4 waves

  dim3 gProj(DM / 64, rowsTot / 64, 1);          // (16, 64)
  k_proj_gemm<<<gProj, blk, 0, stream>>>(inQ, WQ, Qf);
  k_proj_gemm<<<gProj, blk, 0, stream>>>(inK, WK, Kf);
  k_proj_gemm<<<gProj, blk, 0, stream>>>(inV, WV, Vf);

  dim3 gScore(SEQ / 64, SEQ / 64, BATCH * NHEADS);  // (32, 32, 32)
  k_scores<<<gScore, blk, 0, stream>>>(Qf, Kf, out_attn);

  mha_softmax<<<BATCH * NHEADS * SEQ, 256, 0, stream>>>(out_attn, mask);

  dim3 gCtx(1, SEQ / 64, BATCH * NHEADS);           // (1, 32, 32)
  k_context<<<gCtx, blk, 0, stream>>>(out_attn, Vf, Ctx);

  k_outproj<<<gProj, blk, 0, stream>>>(Ctx, Wfc, proj);

  mha_ln_residual<<<BATCH * SEQ, 256, 0, stream>>>(proj, inQ, out_norm);
}
